// FoldableBlock4D_Train_48146583388187
// MI455X (gfx1250) — matrix operand order repaired and checked by reading, not compiled, not run on hardware
//
#include <hip/hip_runtime.h>
#include <hip/hip_bf16.h>

// ---------------------------------------------------------------------------
// FoldableBlock4D: LayerNorm2d + depthwise-CPE residual + windowed 5x5
// neighborhood attention (2 branches x 4 heads, hd=32) + lambda branch join.
//
// GEMM parts (q/k/v proj, output proj) run on v_wmma_f32_16x16x32_f16;
// the 25-tap neighborhood softmax is VALU from LDS. Branch-join coefficients
// are folded into the output-projection A-fragments so one f32 LDS
// accumulator holds the joined result directly. Wave index is forced scalar
// (readfirstlane) so every WMMA executes in uniform control flow (EXEC all-1).
// ---------------------------------------------------------------------------

#define DIMC   128
#define HEADS  4
#define HD     32
#define BR     2
#define WSZ    16
#define NPIX   256          // WSZ*WSZ
#define HW     64
#define B_IMG  4
#define LAMB   0.5f
#define EPSLN  1e-6f
#define QSCALE 0.1767766952966369f   // 1/sqrt(32)

#define XWT_LD 136          // 128 + 8 pad (halfs) -> breaks LDS bank period
#define OT_LD  40           // 32 + 8 pad (halfs)

// LDS partition (bytes)
#define OFF_XWT 0
#define OFF_Q   69632                       // 256*136*2
#define OFF_K   (OFF_Q + 16384)             // 32*256*2
#define OFF_V   (OFF_K + 16384)
#define OFF_OT  (OFF_V + 16384)
#define OFF_HO  (OFF_OT + 20480)            // 256*40*2
#define SMEM_BYTES (OFF_HO + 131072)        // + 128*256*4  = 270336

typedef __attribute__((ext_vector_type(16))) _Float16 v16h;
typedef __attribute__((ext_vector_type(8)))  float    v8f;

#define WMMA_F16(a, b, c) \
  __builtin_amdgcn_wmma_f32_16x16x32_f16(false, (a), false, (b), (short)0, (c), false, false)

// --- A fragment (16x32 f16, M x K) from row-major f32 matrix, scaled ---
// Layout (ISA 7.12.2): lane L holds row M = m0 + (L&15);
// VGPR v, element e: K = k0 + (v/4)*16 + (L>>4)*8 + 2*(v&3) + e.
__device__ __forceinline__ v16h load_a_f32(const float* __restrict__ M, int ld,
                                           int m0, int k0, int lane, float scl) {
  const float* row = M + (size_t)(m0 + (lane & 15)) * ld + k0 + ((lane >> 4) << 3);
  v16h a;
#pragma unroll
  for (int v = 0; v < 8; ++v) {
    const int k = ((v >> 2) << 4) + ((v & 3) << 1);
    a[2 * v]     = (_Float16)(row[k] * scl);
    a[2 * v + 1] = (_Float16)(row[k + 1] * scl);
  }
  return a;
}

// --- B fragment (32x16 f16, K x N) from LDS stored K-contiguous per column:
// BT[n*ldk + k]. Lane L -> column N = n0 + (L&15), K = k0 + (e/8)*16 + (L>>4)*8 + (e&7).
__device__ __forceinline__ v16h load_b_ldsT(const _Float16* __restrict__ BT, int ldk,
                                            int n0, int k0, int lane) {
  const _Float16* col = BT + (size_t)(n0 + (lane & 15)) * ldk + k0 + ((lane >> 4) << 3);
  v16h b;
#pragma unroll
  for (int e = 0; e < 16; ++e) b[e] = col[((e >> 3) << 4) + (e & 7)];
  return b;
}

// --- C/D (16x16 f32) <-> LDS row-major [M][ld]: lane L -> N = n0 + (L&15),
// VGPR r -> M = m0 + 8*(L>>4) + r.
__device__ __forceinline__ v8f load_c_f(const float* __restrict__ C, int ld,
                                        int m0, int n0, int lane) {
  const int n = n0 + (lane & 15);
  const int mb = m0 + ((lane >> 4) << 3);
  v8f c;
#pragma unroll
  for (int r = 0; r < 8; ++r) c[r] = C[(size_t)(mb + r) * ld + n];
  return c;
}

__device__ __forceinline__ void store_c_f(float* __restrict__ C, int ld,
                                          int m0, int n0, v8f c, int lane) {
  const int n = n0 + (lane & 15);
  const int mb = m0 + ((lane >> 4) << 3);
#pragma unroll
  for (int r = 0; r < 8; ++r) C[(size_t)(mb + r) * ld + n] = c[r];
}

__device__ __forceinline__ void store_d_h(_Float16* __restrict__ D, int ld,
                                          int m0, int n0, v8f d, int lane) {
  const int n = n0 + (lane & 15);
  const int mb = m0 + ((lane >> 4) << 3);
#pragma unroll
  for (int r = 0; r < 8; ++r) D[(size_t)(mb + r) * ld + n] = (_Float16)d[r];
}

// ---------------------------------------------------------------------------
// Kernel 1: LayerNorm2d over channels, one pixel per thread.
// ---------------------------------------------------------------------------
__global__ __launch_bounds__(256) void ln_kernel(const float* __restrict__ x,
                                                 const float* __restrict__ g,
                                                 const float* __restrict__ bta,
                                                 float* __restrict__ y) {
  const int p = blockIdx.x * 256 + threadIdx.x;     // 0 .. B*H*W-1
  if (p >= B_IMG * HW * HW) return;
  const int b  = p / (HW * HW);
  const int hw = p - b * HW * HW;
  const float* xb = x + (size_t)b * DIMC * HW * HW + hw;
  float s = 0.f, ss = 0.f;
#pragma unroll 4
  for (int c = 0; c < DIMC; ++c) {
    const float v = xb[(size_t)c * HW * HW];
    s += v; ss += v * v;
  }
  const float mean = s * (1.0f / DIMC);
  float var = ss * (1.0f / DIMC) - mean * mean;
  var = var < 0.f ? 0.f : var;
  const float rstd = rsqrtf(var + EPSLN);
  float* yb = y + (size_t)b * DIMC * HW * HW + hw;
#pragma unroll 4
  for (int c = 0; c < DIMC; ++c) {
    const float v = xb[(size_t)c * HW * HW];
    yb[(size_t)c * HW * HW] = (v - mean) * rstd * g[c] + bta[c];
  }
}

// ---------------------------------------------------------------------------
// Kernel 2: y2 = yln + depthwise3x3_replicate(yln) + bias
// ---------------------------------------------------------------------------
__global__ __launch_bounds__(256) void cpe_kernel(const float* __restrict__ yln,
                                                  const float* __restrict__ w,
                                                  const float* __restrict__ bias,
                                                  float* __restrict__ y2) {
  const int idx = blockIdx.x * 256 + threadIdx.x;
  if (idx >= B_IMG * DIMC * HW * HW) return;
  const int wc = idx & (HW - 1);
  const int hr = (idx >> 6) & (HW - 1);
  const int c  = (idx >> 12) & (DIMC - 1);
  const int b  = idx >> 19;
  const float* plane = yln + (size_t)(b * DIMC + c) * HW * HW;
  float acc = 0.f;
#pragma unroll
  for (int kh = 0; kh < 3; ++kh) {
#pragma unroll
    for (int kw = 0; kw < 3; ++kw) {
      int hh = hr + kh - 1; hh = hh < 0 ? 0 : (hh > HW - 1 ? HW - 1 : hh);
      int ww = wc + kw - 1; ww = ww < 0 ? 0 : (ww > HW - 1 ? HW - 1 : ww);
      acc += w[c * 9 + kh * 3 + kw] * plane[hh * HW + ww];
    }
  }
  y2[idx] = plane[hr * HW + wc] + acc + bias[c];
}

// ---------------------------------------------------------------------------
// Kernel 3: one block per 16x16 window (64 blocks, 256 threads = 8 wave32).
// ---------------------------------------------------------------------------
__global__ __launch_bounds__(256) void attn_kernel(
    const float* __restrict__ y2, const float* __restrict__ Wq,
    const float* __restrict__ Wk, const float* __restrict__ Wv,
    const float* __restrict__ Wo, const float* __restrict__ x,
    float* __restrict__ out) {
  extern __shared__ __align__(16) char smem[];
  _Float16* xwT = (_Float16*)(smem + OFF_XWT);   // [256 pix][136] f16 (c-contig)
  _Float16* Qs  = (_Float16*)(smem + OFF_Q);     // [32 d][256 pix]
  _Float16* Ks  = (_Float16*)(smem + OFF_K);
  _Float16* Vs  = (_Float16*)(smem + OFF_V);
  _Float16* oT  = (_Float16*)(smem + OFF_OT);    // [256 pix][40] (d-contig)
  float*    hoS = (float*)  (smem + OFF_HO);     // [128 c][256 pix] joined acc

  const int tid  = threadIdx.x;
  const int lane = tid & 31;
  // Force wave index into an SGPR: guarantees uniform control flow / scalar
  // addressing around every WMMA (EXEC stays all-1s, per ISA requirement).
  const int wave = __builtin_amdgcn_readfirstlane(tid >> 5);
  const int n    = blockIdx.x;
  const int bimg = n >> 4;          // nh*nw = 16
  const int wh   = (n >> 2) & 3;
  const int wwi  = n & 3;

  // stage window (pixel-major, K-contiguous) + zero joined accumulator
  for (int idx = tid; idx < DIMC * NPIX; idx += 256) {
    const int c = idx >> 8, p = idx & 255;
    const int hh = wh * WSZ + (p >> 4), gw = wwi * WSZ + (p & 15);
    xwT[p * XWT_LD + c] =
        (_Float16)y2[(((size_t)bimg * DIMC + c) * HW + hh) * HW + gw];
  }
  for (int idx = tid; idx < DIMC * NPIX; idx += 256) hoS[idx] = 0.f;
  __syncthreads();

  const float inv_rect = 0.8944271909999159f;   // 1/sqrt(1 + LAM^2*(BR-1))

  for (int bh = 0; bh < BR * HEADS; ++bh) {
    const int br = bh >> 2;

    // ---- q/k/v projections: [32x128] x [128x256] per matrix ----
    const float* Wmats[3] = {Wq + (size_t)bh * HD * DIMC,
                             Wk + (size_t)bh * HD * DIMC,
                             Wv + (size_t)bh * HD * DIMC};
    _Float16* Dsts[3] = {Qs, Ks, Vs};
#pragma unroll
    for (int mat = 0; mat < 3; ++mat) {
#pragma unroll
      for (int mt = 0; mt < 2; ++mt) {
        v16h afr[4];
#pragma unroll
        for (int ks = 0; ks < 4; ++ks)
          afr[ks] = load_a_f32(Wmats[mat], DIMC, mt * 16, ks * 32, lane, 1.0f);
        // wave w handles N-tiles {w, w+8}: fixed trip count, uniform branches
#pragma unroll
        for (int s = 0; s < 2; ++s) {
          const int nt = wave + 8 * s;
          v8f acc = {0.f, 0.f, 0.f, 0.f, 0.f, 0.f, 0.f, 0.f};
#pragma unroll
          for (int ks = 0; ks < 4; ++ks) {
            v16h bfr = load_b_ldsT(xwT, XWT_LD, nt * 16, ks * 32, lane);
            acc = WMMA_F16(afr[ks], bfr, acc);
          }
          store_d_h(Dsts[mat], NPIX, mt * 16, nt * 16, acc, lane);
        }
      }
    }
    __syncthreads();

    // ---- 5x5 neighborhood softmax-attention, one pixel per thread ----
    {
      const int p = tid, pi = p >> 4, pj = p & 15;
      float qv[HD];
#pragma unroll
      for (int d = 0; d < HD; ++d) qv[d] = (float)Qs[d * NPIX + p] * QSCALE;
      float lg[25];
      float mx = -1e30f;
#pragma unroll
      for (int nn = 0; nn < 25; ++nn) {
        const int ii = pi + nn / 5 - 2, jj = pj + nn % 5 - 2;
        float l = 0.f;                 // zero-padded k => logit exactly 0 OOB
        if (ii >= 0 && ii < WSZ && jj >= 0 && jj < WSZ) {
          const int pn = ii * WSZ + jj;
#pragma unroll
          for (int d = 0; d < HD; ++d) l += qv[d] * (float)Ks[d * NPIX + pn];
        }
        lg[nn] = l;
        mx = fmaxf(mx, l);
      }
      float denom = 0.f;
#pragma unroll
      for (int nn = 0; nn < 25; ++nn) { lg[nn] = __expf(lg[nn] - mx); denom += lg[nn]; }
      const float inv = 1.f / denom;
      float ov[HD];
#pragma unroll
      for (int d = 0; d < HD; ++d) ov[d] = 0.f;
#pragma unroll
      for (int nn = 0; nn < 25; ++nn) {
        const int ii = pi + nn / 5 - 2, jj = pj + nn % 5 - 2;
        if (ii >= 0 && ii < WSZ && jj >= 0 && jj < WSZ) {
          const int pn = ii * WSZ + jj;
          const float wn = lg[nn] * inv;
#pragma unroll
          for (int d = 0; d < HD; ++d) ov[d] += wn * (float)Vs[d * NPIX + pn];
        }
      }
#pragma unroll
      for (int d = 0; d < HD; ++d) oT[p * OT_LD + d] = (_Float16)ov[d];
    }
    __syncthreads();

    // ---- output projection [128x32] x [32x256], branch-join coef folded
    //      into A; wave w owns rows [16w,16w+16) of the joined accumulator ----
    {
      const float coef = (br == 0 ? 1.0f : LAMB) * inv_rect;
      const int m0 = wave * 16;
      const v16h a = load_a_f32(Wo + (size_t)bh * DIMC * HD, HD, m0, 0, lane, coef);
#pragma unroll 4
      for (int nt = 0; nt < 16; ++nt) {
        v8f c = load_c_f(hoS, NPIX, m0, nt * 16, lane);
        v16h bfr = load_b_ldsT(oT, OT_LD, nt * 16, 0, lane);
        c = WMMA_F16(a, bfr, c);
        store_c_f(hoS, NPIX, m0, nt * 16, c, lane);
      }
    }
    __syncthreads();
  }

  // epilogue: window reverse + residual
  for (int idx = tid; idx < DIMC * NPIX; idx += 256) {
    const int c = idx >> 8, p = idx & 255;
    const int hh = wh * WSZ + (p >> 4), gw = wwi * WSZ + (p & 15);
    const size_t g = (((size_t)bimg * DIMC + c) * HW + hh) * HW + gw;
    out[g] = x[g] + hoS[c * NPIX + p];
  }
}

// ---------------------------------------------------------------------------
extern "C" void kernel_launch(void* const* d_in, const int* in_sizes, int n_in,
                              void* d_out, int out_size, void* d_ws, size_t ws_size,
                              hipStream_t stream) {
  (void)in_sizes; (void)n_in; (void)out_size; (void)ws_size;
  const float* x   = (const float*)d_in[0];
  const float* Wq  = (const float*)d_in[1];
  const float* Wk  = (const float*)d_in[2];
  const float* Wv  = (const float*)d_in[3];
  const float* Wo  = (const float*)d_in[4];
  const float* cw  = (const float*)d_in[5];
  const float* cb  = (const float*)d_in[6];
  const float* lg  = (const float*)d_in[7];
  const float* lb  = (const float*)d_in[8];
  float* out = (float*)d_out;

  // workspace: yln (8 MB) | y2 (8 MB)
  float* yln = (float*)d_ws;
  float* y2  = yln + (size_t)B_IMG * DIMC * HW * HW;

  ln_kernel<<<(B_IMG * HW * HW) / 256, 256, 0, stream>>>(x, lg, lb, yln);
  cpe_kernel<<<(B_IMG * DIMC * HW * HW) / 256, 256, 0, stream>>>(yln, cw, cb, y2);

  (void)hipFuncSetAttribute(reinterpret_cast<const void*>(attn_kernel),
                            hipFuncAttributeMaxDynamicSharedMemorySize, SMEM_BYTES);
  attn_kernel<<<B_IMG * 16, 256, SMEM_BYTES, stream>>>(y2, Wq, Wk, Wv, Wo, x, out);
}
